// GraphAttentionLayer_82952998355875
// MI455X (gfx1250) — matrix operand order repaired and checked by reading, not compiled, not run on hardware
//
#include <hip/hip_runtime.h>
#include <hip/hip_bf16.h>

// GAT layer fused for MI455X (gfx1250, wave32, WMMA).
// Needs ws_size >= ~6.6 MB scratch (Wh f32 4MB + WhT f16 2MB + WT f16 128KB + e1/e2 256KB + maxe2).

#define B_  4
#define N_  1024
#define E_  256
#define H_  8
#define DH_ 32
#define ALPHA_ 0.2f

typedef __attribute__((ext_vector_type(16))) _Float16 v16h;
typedef __attribute__((ext_vector_type(8)))  float    v8f;

// ---------------------------------------------------------------------------
// Kernel 0: WT[n][k] = (f16) W[k][n]  (so GEMM B-operand is K-contiguous/lane)
// ---------------------------------------------------------------------------
__global__ __launch_bounds__(256) void prep_wt_kernel(const float* __restrict__ W,
                                                      _Float16* __restrict__ WT) {
  int idx = blockIdx.x * 256 + threadIdx.x;          // 65536 elements
  int n = idx >> 8, k = idx & 255;
  WT[(size_t)n * E_ + k] = (_Float16)W[(size_t)k * E_ + n];
}

// ---------------------------------------------------------------------------
// Kernel 1: Wh = x @ W + b_W via v_wmma_f32_16x16x32_f16.
// Writes Wh (fp32, for e12) and WhT (fp16, [B,E,N], for the AV GEMM B-operand).
// One wave per 16x16 output tile; 8 waves/block. K = 256 -> 8 WMMA / tile.
// ---------------------------------------------------------------------------
__global__ __launch_bounds__(256) void gemm_wh_kernel(const float* __restrict__ x,
                                                      const _Float16* __restrict__ WT,
                                                      const float* __restrict__ bW,
                                                      float* __restrict__ Wh,
                                                      _Float16* __restrict__ WhT) {
  const int tid  = threadIdx.x;
  const int wave = tid >> 5;
  const int lane = tid & 31;
  const int i    = lane & 15;   // row within tile (A frag) == col n (B/C frag)
  const int hi   = lane >> 4;   // lane half selects K sub-range
  const int tile = blockIdx.x * 8 + wave;   // 4096 tiles total
  const int rowtile = tile >> 4;            // 0..255  (B*N/16)
  const int coltile = tile & 15;            // 0..15   (E/16)
  const int row = rowtile * 16 + i;
  const int col = coltile * 16 + i;

  const float*    xrow  = x  + (size_t)row * E_;
  const _Float16* wtrow = WT + (size_t)col * E_;

  v8f acc = {};
  const int cb = hi * 8;
  for (int kt = 0; kt < E_ / 32; ++kt) {
    const int kb = kt * 32;
    v16h af;
    // A-frag layout (contiguous, matching B): lanes<16 K=0..15,
    // lanes>=16 K=16..31; element t -> K = t + hi*16.
#pragma unroll
    for (int g = 0; g < 4; ++g) {
      const int coff = cb * 2 + g * 4;   // 0,4,8,12 (+16 for hi)
      float4 xv = *(const float4*)(xrow + kb + coff);
      af[g * 4 + 0] = (_Float16)xv.x;
      af[g * 4 + 1] = (_Float16)xv.y;
      af[g * 4 + 2] = (_Float16)xv.z;
      af[g * 4 + 3] = (_Float16)xv.w;
    }
    // B-frag: lane n, K-contiguous 16 halfs (lo lanes K=kb..kb+15, hi lanes +16)
    v16h bf = *(const v16h*)(wtrow + kb + hi * 16);
    acc = __builtin_amdgcn_wmma_f32_16x16x32_f16(false, af, false, bf,
                                                 (short)0, acc, false, false);
  }
  const float bias = bW[col];
  // C/D layout: VGPR r holds row M = r + 8*hi, col n = lane&15.
#pragma unroll
  for (int r = 0; r < 8; ++r) {
    const int M = r + 8 * hi;
    const int grow = rowtile * 16 + M;
    const float v = acc[r] + bias;
    Wh[(size_t)grow * E_ + col] = v;
    const int bb = grow >> 10, nloc = grow & 1023;
    WhT[((size_t)bb * E_ + col) * N_ + nloc] = (_Float16)v;
  }
}

// ---------------------------------------------------------------------------
// Kernel 2: e12 = Wh @ a_w + b_a, scattered into e1[b*H+h][n] (c=2h) and
// e2[b*H+h][n] (c=2h+1). Block = 16 rows x 16 cols.
// ---------------------------------------------------------------------------
__global__ __launch_bounds__(256) void e12_kernel(const float* __restrict__ Wh,
                                                  const float* __restrict__ aw,
                                                  const float* __restrict__ ba,
                                                  float* __restrict__ e1,
                                                  float* __restrict__ e2) {
  const int tid = threadIdx.x;
  const int r = tid >> 4, c = tid & 15;
  const int grow = blockIdx.x * 16 + r;            // 0..4095
  const float* whrow = Wh + (size_t)grow * E_;
  float acc = ba[c];
  for (int k = 0; k < E_; ++k) acc += whrow[k] * aw[k * 16 + c];
  const int bb = grow >> 10, nloc = grow & 1023;
  const int h = c >> 1;
  float* dst = (c & 1) ? e2 : e1;
  dst[((size_t)bb * H_ + h) * N_ + nloc] = acc;
}

// ---------------------------------------------------------------------------
// Kernel 3: maxe2[bh] = max_j e2[bh][j]  (upper bound for the softmax shift)
// ---------------------------------------------------------------------------
__global__ __launch_bounds__(256) void maxe2_kernel(const float* __restrict__ e2,
                                                    float* __restrict__ maxe2) {
  __shared__ float red[256];
  const int bh = blockIdx.x;
  float m = -1e30f;
  for (int j = threadIdx.x; j < N_; j += 256)
    m = fmaxf(m, e2[(size_t)bh * N_ + j]);
  red[threadIdx.x] = m;
  __syncthreads();
  for (int s = 128; s > 0; s >>= 1) {
    if (threadIdx.x < s) red[threadIdx.x] = fmaxf(red[threadIdx.x], red[threadIdx.x + s]);
    __syncthreads();
  }
  if (threadIdx.x == 0) maxe2[bh] = red[0];
}

// ---------------------------------------------------------------------------
// Kernel 4: fused mask + softmax + att@Whh + ELU. One pass over A.
// Grid: B*(N/16) blocks; 8 waves/block, wave = head (all heads share A tile).
// Per 32-j tile: build p = A * exp(lrelu(e1+e2) - m) directly in the WMMA
// A-fragment layout, 2x v_wmma_f32_16x16x32_f16 into the DH=32 accumulator.
// ---------------------------------------------------------------------------
__global__ __launch_bounds__(256) void gat_attn_kernel(const float* __restrict__ A,
                                                       const _Float16* __restrict__ WhT,
                                                       const float* __restrict__ e1g,
                                                       const float* __restrict__ e2g,
                                                       const float* __restrict__ maxe2,
                                                       float* __restrict__ out) {
  const int tid  = threadIdx.x;
  const int h    = tid >> 5;           // wave index == head
  const int lane = tid & 31;
  const int i    = lane & 15;          // tile row (A frag) == output col n
  const int hi   = lane >> 4;
  const int blk  = blockIdx.x;         // b*64 + itile
  const int b    = blk >> 6;
  const int ibase = (blk & 63) << 4;
  const int bh   = b * H_ + h;

  const float e1v = e1g[(size_t)bh * N_ + ibase + i];
  float m = e1v + maxe2[bh];
  m = (m > 0.f) ? m : ALPHA_ * m;      // leakyrelu of an upper bound >= all masked scores

  const float*    Arow  = A   + ((size_t)(b * N_ + ibase + i)) * N_;
  const float*    e2row = e2g + (size_t)bh * N_;
  const _Float16* Bp0   = WhT + ((size_t)b * E_ + h * DH_ + i) * N_;        // d = i
  const _Float16* Bp1   = WhT + ((size_t)b * E_ + h * DH_ + 16 + i) * N_;   // d = 16+i

  v8f acc0 = {}, acc1 = {};
  float psum = 0.f;
  const int cb = hi * 8;

  for (int jt = 0; jt < N_ / 32; ++jt) {
    const int jbase = jt * 32;
    v16h ap;
#pragma unroll
    for (int g = 0; g < 4; ++g) {
      const int coff = cb * 2 + g * 4;
      float4 av = *(const float4*)(Arow  + jbase + coff);
      float4 ev = *(const float4*)(e2row + jbase + coff);
      float s, p;
      s = e1v + ev.x; s = (s > 0.f) ? s : ALPHA_ * s;
      p = (av.x != 0.f) ? __expf(s - m) : 0.f; psum += p; ap[g * 4 + 0] = (_Float16)p;
      s = e1v + ev.y; s = (s > 0.f) ? s : ALPHA_ * s;
      p = (av.y != 0.f) ? __expf(s - m) : 0.f; psum += p; ap[g * 4 + 1] = (_Float16)p;
      s = e1v + ev.z; s = (s > 0.f) ? s : ALPHA_ * s;
      p = (av.z != 0.f) ? __expf(s - m) : 0.f; psum += p; ap[g * 4 + 2] = (_Float16)p;
      s = e1v + ev.w; s = (s > 0.f) ? s : ALPHA_ * s;
      p = (av.w != 0.f) ? __expf(s - m) : 0.f; psum += p; ap[g * 4 + 3] = (_Float16)p;
    }
    v16h b0 = *(const v16h*)(Bp0 + jbase + hi * 16);
    v16h b1 = *(const v16h*)(Bp1 + jbase + hi * 16);
    acc0 = __builtin_amdgcn_wmma_f32_16x16x32_f16(false, ap, false, b0,
                                                  (short)0, acc0, false, false);
    acc1 = __builtin_amdgcn_wmma_f32_16x16x32_f16(false, ap, false, b1,
                                                  (short)0, acc1, false, false);
    if (jt + 1 < N_ / 32)
      __builtin_prefetch(Arow + jbase + 32, 0, 0);   // -> global_prefetch_b8
  }

  // denom[i] = psum(lane i) + psum(lane i+16); self-loops guarantee > 0.
#pragma unroll
  for (int r = 0; r < 8; ++r) {
    const int M = r + 8 * hi;
    const float denom = __shfl(psum, M, 32) + __shfl(psum, M + 16, 32);
    const float inv = 1.f / denom;
    float o0 = acc0[r] * inv;
    float o1 = acc1[r] * inv;
    o0 = (o0 > 0.f) ? o0 : (__expf(o0) - 1.f);       // ELU (alpha=1)
    o1 = (o1 > 0.f) ? o1 : (__expf(o1) - 1.f);
    float* orow = out + ((size_t)(b * N_ + ibase + M)) * E_ + h * DH_;
    orow[i]      = o0;
    orow[16 + i] = o1;
  }
}

// ---------------------------------------------------------------------------
extern "C" void kernel_launch(void* const* d_in, const int* in_sizes, int n_in,
                              void* d_out, int out_size, void* d_ws, size_t ws_size,
                              hipStream_t stream) {
  const float* x  = (const float*)d_in[0];   // [B,N,E]
  const float* A  = (const float*)d_in[1];   // [B,N,N]
  const float* W  = (const float*)d_in[2];   // [E,E]
  const float* bW = (const float*)d_in[3];   // [E]
  const float* aw = (const float*)d_in[4];   // [E,2H]
  const float* ba = (const float*)d_in[5];   // [2H]
  float* out = (float*)d_out;                // [B,N,E]

  char* ws = (char*)d_ws;
  float*    Wh  = (float*)(ws);                                   // 4 MB
  _Float16* WhT = (_Float16*)(ws + (4u << 20));                   // 2 MB
  _Float16* WT  = (_Float16*)(ws + (6u << 20));                   // 128 KB
  float*    e1  = (float*)(ws + (6u << 20) + (128u << 10));       // 128 KB
  float*    e2  = (float*)(ws + (6u << 20) + (256u << 10));       // 128 KB
  float*    mx  = (float*)(ws + (6u << 20) + (384u << 10));       // 128 B

  prep_wt_kernel <<<256, 256, 0, stream>>>(W, WT);
  gemm_wh_kernel <<<512, 256, 0, stream>>>(x, WT, bW, Wh, WhT);
  e12_kernel     <<<256, 256, 0, stream>>>(Wh, aw, ba, e1, e2);
  maxe2_kernel   <<< 32, 256, 0, stream>>>(e2, mx);
  gat_attn_kernel<<<256, 256, 0, stream>>>(A, WhT, e1, e2, mx, out);
}
